// DotProductAttention_80874234183906
// MI455X (gfx1250) — matrix operand order repaired and checked by reading, not compiled, not run on hardware
//
#include <hip/hip_runtime.h>
#include <math.h>

// MI455X / gfx1250, wave32. All matrix math via v_wmma_f32_16x16x32_f16.
// Q/K/V are materialized once as packed f16 (numerically identical to
// converting per-use), so the score GEMM loads Q/K WMMA fragments directly
// with b128 loads and zero conversion ALU; AV loads V as strided f16, no convert.
typedef __attribute__((ext_vector_type(16))) _Float16 v16h;
typedef __attribute__((ext_vector_type(8)))  _Float16 v8h;
typedef __attribute__((ext_vector_type(8)))  float    v8f;
typedef __attribute__((ext_vector_type(4)))  float    f32x4;

#define S_LEN   2048
#define D_MODEL 1024
#define D_K     128
#define BATCH   4

// ---- WMMA fragment helpers (wave32; A/B take K contiguously, k = 16*(lane/16) + i) ----

// A fragment from a row-major fp32 matrix (convert on the fly).
// lane<16: row M=lane, halves 0..15 = K k0..k0+15
// lane>=16: row M=lane-16, halves 0..15 = K k0+16..k0+31
static __device__ __forceinline__ v16h load_a_f32(const float* __restrict__ A, int lda,
                                                  int row0, int k0, int lane) {
  int row = row0 + (lane & 15);
  int kb  = k0 + ((lane >> 4) << 4);
  const float* p = A + (size_t)row * lda + kb;
  f32x4 x0 = *(const f32x4*)(p);
  f32x4 x1 = *(const f32x4*)(p + 4);
  f32x4 x2 = *(const f32x4*)(p + 8);
  f32x4 x3 = *(const f32x4*)(p + 12);
  v16h a;
#pragma unroll
  for (int i = 0; i < 4; ++i) {
    a[i]      = (_Float16)x0[i];
    a[4 + i]  = (_Float16)x1[i];
    a[8 + i]  = (_Float16)x2[i];
    a[12 + i] = (_Float16)x3[i];
  }
  return a;
}

// A fragment from a row-major packed-f16 matrix: two b128 loads, no ALU.
static __device__ __forceinline__ v16h load_a_f16(const _Float16* __restrict__ A, int lda,
                                                  int row0, int k0, int lane) {
  int row = row0 + (lane & 15);
  int kb  = k0 + ((lane >> 4) << 4);
  const _Float16* p = A + (size_t)row * lda + kb;
  v8h lo = *(const v8h*)(p);        // K kb..kb+7
  v8h hi = *(const v8h*)(p + 8);    // K kb+8..kb+15
  v16h a;
#pragma unroll
  for (int i = 0; i < 8; ++i) { a[i] = lo[i]; a[8 + i] = hi[i]; }
  return a;
}

// B fragment where memory holds B^T row-major f16 (contiguous along K):
// B[k][n] = Bt[n*ldb + k]. lane<16: col N=lane, halves 0..15 = K k0..k0+15;
// lane>=16: K k0+16..k0+31.  Two b128 loads, no ALU.
static __device__ __forceinline__ v16h load_b_nmajor_f16(const _Float16* __restrict__ Bt, int ldb,
                                                         int col0, int k0, int lane) {
  int col = col0 + (lane & 15);
  int kb  = k0 + ((lane >> 4) << 4);
  const _Float16* p = Bt + (size_t)col * ldb + kb;
  v8h lo = *(const v8h*)(p);
  v8h hi = *(const v8h*)(p + 8);
  v16h b;
#pragma unroll
  for (int i = 0; i < 8; ++i) { b[i] = lo[i]; b[8 + i] = hi[i]; }
  return b;
}

// B fragment from a row-major f16 [K x N] matrix (strided d16 loads).
static __device__ __forceinline__ v16h load_b_kmajor_f16(const _Float16* __restrict__ B, int ldb,
                                                         int col0, int k0, int lane) {
  int col = col0 + (lane & 15);
  int kb  = k0 + ((lane >> 4) << 4);
  const _Float16* p = B + (size_t)kb * ldb + col;
  v16h b;
#pragma unroll
  for (int i = 0; i < 16; ++i) b[i] = p[(size_t)i * ldb];
  return b;
}

// B fragment from a row-major fp32 [K x N] matrix (strided, convert).
static __device__ __forceinline__ v16h load_b_kmajor_f32(const float* __restrict__ B, int ldb,
                                                         int col0, int k0, int lane) {
  int col = col0 + (lane & 15);
  int kb  = k0 + ((lane >> 4) << 4);
  const float* p = B + (size_t)kb * ldb + col;
  v16h b;
#pragma unroll
  for (int i = 0; i < 16; ++i) b[i] = (_Float16)p[(size_t)i * ldb];
  return b;
}

// C/D: VGPR r holds M = r (lanes 0-15) or M = 8+r (lanes 16-31), N = lane&15.
static __device__ __forceinline__ void store_c_f32(float* __restrict__ C, int ldc,
                                                   int row0, int col0, int lane, v8f c) {
  int n     = col0 + (lane & 15);
  int mbase = row0 + ((lane >> 4) << 3);
#pragma unroll
  for (int r = 0; r < 8; ++r) C[(size_t)(mbase + r) * ldc + n] = c[r];
}

static __device__ __forceinline__ void store_c_f16(_Float16* __restrict__ C, int ldc,
                                                   int row0, int col0, int lane, v8f c) {
  int n     = col0 + (lane & 15);
  int mbase = row0 + ((lane >> 4) << 3);
#pragma unroll
  for (int r = 0; r < 8; ++r) C[(size_t)(mbase + r) * ldc + n] = (_Float16)c[r];
}

// ---- Kernel 1: Q/K/V projections.  [8192,1024] x [1024,128] -> packed f16.
// Each wave computes a 64x16 output slab (4 M-tiles): the strided weight
// fragment is loaded once per K-chunk and reused by 4 WMMAs.
__global__ void __launch_bounds__(32)
dpa_qkv_proj(const float* __restrict__ x, const float* __restrict__ wq,
             const float* __restrict__ wk, const float* __restrict__ wv,
             _Float16* __restrict__ Q, _Float16* __restrict__ K, _Float16* __restrict__ V) {
  const int lane  = threadIdx.x;
  const int row0  = blockIdx.x * 64;            // 128 M-blocks over B*S
  const int col0  = blockIdx.y * 16;            // 8 N-tiles over D_K
  const int which = blockIdx.z;                 // 0=Q 1=K 2=V
  const float* W = (which == 0) ? wq : (which == 1) ? wk : wv;
  _Float16*    O = (which == 0) ? Q  : (which == 1) ? K  : V;
  v8f c0 = {}, c1 = {}, c2 = {}, c3 = {};
  for (int k0 = 0; k0 < D_MODEL; k0 += 32) {
    v16h b  = load_b_kmajor_f32(W, D_K, col0, k0, lane);
    v16h a0 = load_a_f32(x, D_MODEL, row0,      k0, lane);
    v16h a1 = load_a_f32(x, D_MODEL, row0 + 16, k0, lane);
    v16h a2 = load_a_f32(x, D_MODEL, row0 + 32, k0, lane);
    v16h a3 = load_a_f32(x, D_MODEL, row0 + 48, k0, lane);
    c0 = __builtin_amdgcn_wmma_f32_16x16x32_f16(false, a0, false, b, (short)0, c0, false, false);
    c1 = __builtin_amdgcn_wmma_f32_16x16x32_f16(false, a1, false, b, (short)0, c1, false, false);
    c2 = __builtin_amdgcn_wmma_f32_16x16x32_f16(false, a2, false, b, (short)0, c2, false, false);
    c3 = __builtin_amdgcn_wmma_f32_16x16x32_f16(false, a3, false, b, (short)0, c3, false, false);
  }
  store_c_f16(O, D_K, row0,      col0, lane, c0);
  store_c_f16(O, D_K, row0 + 16, col0, lane, c1);
  store_c_f16(O, D_K, row0 + 32, col0, lane, c2);
  store_c_f16(O, D_K, row0 + 48, col0, lane, c3);
}

// ---- Kernel 2: raw scores = Q K^T / sqrt(d_k).  Pure-f16 fragment loads.
// Each wave computes 64 q-rows x 16 k-cols; the K-column fragment is reused 4x.
// Blocks entirely above the causal diagonal are skipped (uniform branch).
__global__ void __launch_bounds__(32)
dpa_scores(const _Float16* __restrict__ Q, const _Float16* __restrict__ K,
           float* __restrict__ attn) {
  const int qb = blockIdx.x, kt = blockIdx.y, b = blockIdx.z;
  const int q0 = qb * 64, k0c = kt * 16;
  if (q0 + 63 < k0c) return;     // whole 64-row block masked; normalize writes the zeros
  const int lane = threadIdx.x;
  const _Float16* Qb = Q + (size_t)b * S_LEN * D_K;
  const _Float16* Kb = K + (size_t)b * S_LEN * D_K;
  float*          Ab = attn + (size_t)b * S_LEN * S_LEN;
  v8f c0 = {}, c1 = {}, c2 = {}, c3 = {};
  for (int d0 = 0; d0 < D_K; d0 += 32) {
    v16h bf = load_b_nmajor_f16(Kb, D_K, k0c, d0, lane);   // contiguous along d
    v16h a0 = load_a_f16(Qb, D_K, q0,      d0, lane);
    v16h a1 = load_a_f16(Qb, D_K, q0 + 16, d0, lane);
    v16h a2 = load_a_f16(Qb, D_K, q0 + 32, d0, lane);
    v16h a3 = load_a_f16(Qb, D_K, q0 + 48, d0, lane);
    c0 = __builtin_amdgcn_wmma_f32_16x16x32_f16(false, a0, false, bf, (short)0, c0, false, false);
    c1 = __builtin_amdgcn_wmma_f32_16x16x32_f16(false, a1, false, bf, (short)0, c1, false, false);
    c2 = __builtin_amdgcn_wmma_f32_16x16x32_f16(false, a2, false, bf, (short)0, c2, false, false);
    c3 = __builtin_amdgcn_wmma_f32_16x16x32_f16(false, a3, false, bf, (short)0, c3, false, false);
  }
  const float scale = 0.08838834764831845f;  // 1/sqrt(128)
#pragma unroll
  for (int r = 0; r < 8; ++r) { c0[r] *= scale; c1[r] *= scale; c2[r] *= scale; c3[r] *= scale; }
  store_c_f32(Ab, S_LEN, q0,      k0c, lane, c0);
  store_c_f32(Ab, S_LEN, q0 + 16, k0c, lane, c1);
  store_c_f32(Ab, S_LEN, q0 + 32, k0c, lane, c2);
  store_c_f32(Ab, S_LEN, q0 + 48, k0c, lane, c3);
}

// ---- Kernel 3: per-key-column max and sum(exp) over q in [k, S).
__global__ void __launch_bounds__(256)
dpa_colstats(const float* __restrict__ attn, float* __restrict__ colmax,
             float* __restrict__ colsum) {
  const int tid = threadIdx.x;
  const int col = blockIdx.x;                // b*S + k
  const int b = col >> 11, k = col & (S_LEN - 1);
  const float* Ab = attn + (size_t)b * S_LEN * S_LEN;
  __shared__ float red[256];
  float m = -INFINITY;
  for (int q = k + tid; q < S_LEN; q += 256) m = fmaxf(m, Ab[(size_t)q * S_LEN + k]);
  red[tid] = m; __syncthreads();
  for (int s = 128; s > 0; s >>= 1) { if (tid < s) red[tid] = fmaxf(red[tid], red[tid + s]); __syncthreads(); }
  m = red[0]; __syncthreads();
  float sum = 0.f;
  for (int q = k + tid; q < S_LEN; q += 256) sum += __expf(Ab[(size_t)q * S_LEN + k] - m);
  red[tid] = sum; __syncthreads();
  for (int s = 128; s > 0; s >>= 1) { if (tid < s) red[tid] += red[tid + s]; __syncthreads(); }
  if (tid == 0) { colmax[col] = m; colsum[col] = red[0]; }
}

// ---- Kernel 4: attn = q>=k ? exp(s - max_k)/sum_k : 0   (writes every element).
__global__ void __launch_bounds__(256)
dpa_normalize(const float* __restrict__ colmax, const float* __restrict__ colsum,
              float* __restrict__ attn) {
  size_t idx = (size_t)blockIdx.x * 256 + threadIdx.x;       // over B*S*S
  int k = (int)(idx & (S_LEN - 1));
  size_t t = idx >> 11;
  int q = (int)(t & (S_LEN - 1));
  int b = (int)(t >> 11);
  float v = 0.f;
  if (q >= k) {
    int c = (b << 11) + k;
    v = __expf(attn[idx] - colmax[c]) / colsum[c];
  }
  attn[idx] = v;
}

// ---- Kernel 5: out = attn @ V.
// Each wave computes 32 q-rows x 16 v-cols (2 M-tiles): the strided f16 V
// fragment is reused 2x; K-loop truncated at the causal boundary of the block.
// attn stays fp32 (it is a kernel output), converted in-register.
__global__ void __launch_bounds__(32)
dpa_av(const float* __restrict__ attn, const _Float16* __restrict__ V,
       float* __restrict__ out) {
  const int lane = threadIdx.x;
  const int qb = blockIdx.x, nt = blockIdx.y, b = blockIdx.z;
  const float*    Ab = attn + (size_t)b * S_LEN * S_LEN;
  const _Float16* Vb = V + (size_t)b * S_LEN * D_K;
  float*          Ob = out + (size_t)b * S_LEN * D_K;
  const int q0 = qb * 32, n0 = nt * 16;
  const int kmax = q0 + 32;  // attn[q][k]==0 for k>q, so over-reading the chunk is safe
  v8f c0 = {}, c1 = {};
  for (int k0 = 0; k0 < kmax; k0 += 32) {
    v16h bf = load_b_kmajor_f16(Vb, D_K, n0, k0, lane);
    v16h a0 = load_a_f32(Ab, S_LEN, q0,      k0, lane);
    v16h a1 = load_a_f32(Ab, S_LEN, q0 + 16, k0, lane);
    c0 = __builtin_amdgcn_wmma_f32_16x16x32_f16(false, a0, false, bf, (short)0, c0, false, false);
    c1 = __builtin_amdgcn_wmma_f32_16x16x32_f16(false, a1, false, bf, (short)0, c1, false, false);
  }
  store_c_f32(Ob, D_K, q0,      n0, lane, c0);
  store_c_f32(Ob, D_K, q0 + 16, n0, lane, c1);
}

extern "C" void kernel_launch(void* const* d_in, const int* in_sizes, int n_in,
                              void* d_out, int out_size, void* d_ws, size_t ws_size,
                              hipStream_t stream) {
  (void)in_sizes; (void)n_in; (void)out_size; (void)ws_size;
  const float* x  = (const float*)d_in[0];   // [4,2048,1024]
  const float* wq = (const float*)d_in[1];   // [1024,128]
  const float* wk = (const float*)d_in[2];
  const float* wv = (const float*)d_in[3];

  // d_out = [out (4*2048*128)] ++ [attention (4*2048*2048)]
  float* out  = (float*)d_out;
  float* attn = out + (size_t)BATCH * S_LEN * D_K;

  // Workspace: Q, K, V packed f16 + column stats (~6.1 MB)
  _Float16* Q = (_Float16*)d_ws;
  _Float16* K = Q + (size_t)BATCH * S_LEN * D_K;
  _Float16* V = K + (size_t)BATCH * S_LEN * D_K;
  float* colmax = (float*)(V + (size_t)BATCH * S_LEN * D_K);
  float* colsum = colmax + (size_t)BATCH * S_LEN;

  dpa_qkv_proj<<<dim3(BATCH * S_LEN / 64, D_K / 16, 3), 32, 0, stream>>>(x, wq, wk, wv, Q, K, V);
  dpa_scores  <<<dim3(S_LEN / 64, S_LEN / 16, BATCH), 32, 0, stream>>>(Q, K, attn);
  dpa_colstats<<<BATCH * S_LEN, 256, 0, stream>>>(attn, colmax, colsum);
  dpa_normalize<<<(unsigned)((size_t)BATCH * S_LEN * S_LEN / 256), 256, 0, stream>>>(colmax, colsum, attn);
  dpa_av      <<<dim3(S_LEN / 32, D_K / 16, BATCH), 32, 0, stream>>>(attn, V, out);
}
